// QSSM_47038481825831
// MI455X (gfx1250) — matrix operand order repaired and checked by reading, not compiled, not run on hardware
//
#include <hip/hip_runtime.h>

// QSSM forward for MI455X (gfx1250): persistent wave32 WMMA kernel with
// async global->LDS staging of shared A operands.
// B=32, T=512, I=512, H=2048, O=512.

#define B_   32
#define T_   512
#define I_   512
#define H_   2048
#define O_   512
#define NWG  32
#define NTHR 256
#define NTOT (NWG * NTHR)   // 8192 threads, 256 waves

// LDS A-slice row strides (halfwords); row stride bytes == 16 mod 256 ->
// 16 lanes (rows) of a half-wave map to banks 4l..4l+3: conflict-free b128.
#define SA_STRIDE 2056              // 2048 + 8 pad  (4112 B/row)
#define SX_STRIDE 520               // 512 + 8 pad   (1040 B/row)
#define SA_BYTES  (16 * SA_STRIDE * 2)   // 65792
#define SX_BYTES  (16 * SX_STRIDE * 2)   // 16640
#define SMEM_BYTES (SA_BYTES + SX_BYTES) // 82432

typedef __attribute__((ext_vector_type(16))) _Float16 v16h;
typedef __attribute__((ext_vector_type(8)))  _Float16 v8h;
typedef __attribute__((ext_vector_type(8)))  float    v8f;

// ---- orderable-uint encoding of float (unsigned order == float order) ----
__device__ __forceinline__ unsigned fenc(float f) {
  unsigned u = __float_as_uint(f);
  return (u & 0x80000000u) ? ~u : (u | 0x80000000u);
}
__device__ __forceinline__ float fdec(unsigned u) {
  unsigned b = (u & 0x80000000u) ? (u & 0x7FFFFFFFu) : ~u;
  return __uint_as_float(b);
}
#define ENC_POS_INF 0xFF800000u  // fenc(+inf)  -> min-slot init
#define ENC_NEG_INF 0x007FFFFFu  // fenc(-inf)  -> max-slot init

// ---- per-tensor 8-bit affine fake-quant (round half-to-even like jnp.round) ----
__device__ __forceinline__ float fq(float x, float mn, float mx) {
  float s = (mx - mn) * (1.0f / 255.0f);
  s = (s == 0.0f) ? 1.0f : s;
  float q = rintf((x - mn) / s);
  q = fminf(fmaxf(q, 0.0f), 255.0f);
  return fmaf(q, s, mn);
}

// ---------------- prep kernels ----------------
__global__ void k_cvt(const float* __restrict__ src, _Float16* __restrict__ dst, int n) {
  int i = blockIdx.x * blockDim.x + threadIdx.x;
  if (i < n) dst[i] = (_Float16)src[i];
}

// per-timestep min/max of x[:, t, :]   (x layout [B][T][I])
__global__ void k_xmm(const float* __restrict__ x, float* __restrict__ xmm) {
  int t = blockIdx.x;
  __shared__ float smn[8], smx[8];
  float mn = __builtin_inff(), mx = -__builtin_inff();
  for (int idx = threadIdx.x; idx < B_ * I_; idx += NTHR) {
    int b = idx >> 9, i = idx & (I_ - 1);
    float v = x[(size_t)b * T_ * I_ + (size_t)t * I_ + i];
    mn = fminf(mn, v);
    mx = fmaxf(mx, v);
  }
  for (int o = 16; o; o >>= 1) {
    mn = fminf(mn, __shfl_xor(mn, o, 32));
    mx = fmaxf(mx, __shfl_xor(mx, o, 32));
  }
  if ((threadIdx.x & 31) == 0) { smn[threadIdx.x >> 5] = mn; smx[threadIdx.x >> 5] = mx; }
  __syncthreads();
  if (threadIdx.x == 0) {
    for (int w = 1; w < 8; ++w) { mn = fminf(mn, smn[w]); mx = fmaxf(mx, smx[w]); }
    xmm[2 * t] = mn;
    xmm[2 * t + 1] = mx;
  }
}

// quantize whole input sequence, re-layout [B][T][I] -> [T][B][I] f16
__global__ void k_xq(const float* __restrict__ x, const float* __restrict__ xmm,
                     _Float16* __restrict__ xq) {
  long long idx = (long long)blockIdx.x * blockDim.x + threadIdx.x;
  if (idx >= (long long)B_ * T_ * I_) return;
  int i = (int)(idx % I_);
  long long r = idx / I_;
  int t = (int)(r % T_);
  int b = (int)(r / T_);
  float q = fq(x[idx], xmm[2 * t], xmm[2 * t + 1]);
  xq[(size_t)t * B_ * I_ + (size_t)b * I_ + i] = (_Float16)q;
}

// zero h0, init min/max slots + barrier state (runs every launch -> replay-safe)
__global__ void k_init(float* __restrict__ hb0, unsigned* __restrict__ hmm,
                       unsigned* __restrict__ bar) {
  int i = blockIdx.x * blockDim.x + threadIdx.x;
  if (i < B_ * H_) hb0[i] = 0.0f;
  if (i == 0) {
    hmm[0] = 0x80000000u;  // fenc(0.0f): h0 min
    hmm[1] = 0x80000000u;  // fenc(0.0f): h0 max
    hmm[2] = ENC_POS_INF;
    hmm[3] = ENC_NEG_INF;
    bar[0] = 0u;           // arrive count (monotonic)
    bar[1] = 0u;           // completed-barrier sense (monotonic)
  }
}

// ---------------- device-wide barrier (all NWG blocks resident) ----------------
__device__ __forceinline__ void gbar(unsigned* cnt, unsigned* sns, unsigned& ph) {
  unsigned want = ++ph;
  __syncthreads();
  __threadfence();
  if (threadIdx.x == 0) {
    unsigned old = atomicAdd(cnt, 1u);
    if (old == want * NWG - 1u) {
      __hip_atomic_store(sns, want, __ATOMIC_RELEASE, __HIP_MEMORY_SCOPE_AGENT);
    } else {
      while (__hip_atomic_load(sns, __ATOMIC_ACQUIRE, __HIP_MEMORY_SCOPE_AGENT) < want)
        __builtin_amdgcn_s_sleep(1);
    }
  }
  __syncthreads();
  __threadfence();
}

// async 16-byte global -> LDS copy (CDNA5; tracked on ASYNCcnt)
__device__ __forceinline__ void async_g2l_b128(unsigned lds_off, const void* gaddr) {
  asm volatile("global_load_async_to_lds_b128 %0, %1, off"
               :: "v"(lds_off), "v"((unsigned long long)(uintptr_t)gaddr)
               : "memory");
}
__device__ __forceinline__ void wait_async0() {
  asm volatile("s_wait_asynccnt 0x0" ::: "memory");
}

// ---------------- persistent scan kernel ----------------
__global__ __launch_bounds__(NTHR, 1) void k_main(
    const _Float16* __restrict__ w16rec,  // [H][H]   f16 (row n = W_rec[n,:])
    const _Float16* __restrict__ w16in,   // [H][I]   f16
    const _Float16* __restrict__ w16out,  // [O][H]   f16
    const _Float16* __restrict__ xq16,    // [T][B][I] f16 (pre-quantized)
    const float* __restrict__ brec,       // [H]
    const float* __restrict__ bout,       // [O]
    _Float16* __restrict__ hq,            // [B][H]   f16 scratch
    float* __restrict__ hb0,              // [B][H]
    float* __restrict__ hb1,              // [B][H]
    unsigned* __restrict__ hmm,           // 2 slots x {min,max} encoded
    unsigned* __restrict__ bar,           // {count, sense}
    float* __restrict__ dout)             // [B][T][O]
{
  extern __shared__ char smem_raw[];     // SA_BYTES + SX_BYTES
  _Float16* sA = (_Float16*)smem_raw;                    // [16][SA_STRIDE]
  _Float16* sX = (_Float16*)(smem_raw + SA_BYTES);       // [16][SX_STRIDE]
  const unsigned sA_off = (unsigned)(uintptr_t)sA;       // LDS byte offsets
  const unsigned sX_off = (unsigned)(uintptr_t)sX;

  const int tid   = threadIdx.x;
  const int lane  = tid & 31;
  const int gwave = blockIdx.x * (NTHR / 32) + (tid >> 5);  // 0..255
  const int gtid  = blockIdx.x * NTHR + tid;                // 0..8191

  // h_new tile ownership: 2 M-tiles x 128 N-tiles = 256 tiles, 1 per wave.
  // All 8 waves of a block share the same M-tile -> A slices staged in LDS.
  const int m0 = (gwave >> 7) * 16;      // uniform within the block
  const int n0 = (gwave & 127) * 16;
  const int l15  = lane & 15;
  const int half = lane >> 4;
  const int kb = half << 4;   // A K-split:      {0..15} / {16..31}
  const int kB = half << 4;   // B K-split:      {0..15} / {16..31}
  const int bN   = n0 + l15;          // B col = H index (row of W_rec / W_in)
  const int mBase = m0 + (half << 3); // C/D: lanes16-31 hold M+8 rows
  const float brv = brec[bN];

  // y tile ownership (waves 0..63): 2 x 32 tiles of [32 x 512]
  const int ymt = gwave >> 5, ynt = gwave & 31;
  const int aRowY = ymt * 16 + l15;
  const int bNY   = ynt * 16 + l15;
  const int mBaseY = ymt * 16 + (half << 3);

  unsigned ph = 0;

  for (int t = 0; t < T_; ++t) {
    const int cur = t & 1, nxt = cur ^ 1;
    const float* hcur = cur ? hb1 : hb0;
    float*       hnx  = cur ? hb0 : hb1;
    const _Float16* xqt = xq16 + (size_t)t * B_ * I_;

    // ---- phase 0: quantize h_t -> f16, reset next min/max slot ----
    {
      float hmn = fdec(__hip_atomic_load(hmm + 2 * cur,     __ATOMIC_RELAXED, __HIP_MEMORY_SCOPE_AGENT));
      float hmx = fdec(__hip_atomic_load(hmm + 2 * cur + 1, __ATOMIC_RELAXED, __HIP_MEMORY_SCOPE_AGENT));
      for (int idx = gtid; idx < B_ * H_; idx += NTOT)
        hq[idx] = (_Float16)fq(hcur[idx], hmn, hmx);
      if (gtid == 0) {
        __hip_atomic_store(hmm + 2 * nxt,     ENC_POS_INF, __ATOMIC_RELAXED, __HIP_MEMORY_SCOPE_AGENT);
        __hip_atomic_store(hmm + 2 * nxt + 1, ENC_NEG_INF, __ATOMIC_RELAXED, __HIP_MEMORY_SCOPE_AGENT);
      }
    }
    gbar(bar, bar + 1, ph);

    // ---- stage A slices (shared by all 8 waves): hq[m0:m0+16, :] and
    //      xq_t[m0:m0+16, :]  global -> LDS via async B128 copies ----
    {
      // hq slice: 16 rows x 256 chunks of 16B
      #pragma unroll
      for (int j = 0; j < 16; ++j) {
        int c = tid + j * NTHR;
        int r = c >> 8, col = c & 255;
        async_g2l_b128(sA_off + (unsigned)(r * (SA_STRIDE * 2) + col * 16),
                       hq + (size_t)(m0 + r) * H_ + col * 8);
      }
      // xq_t slice: 16 rows x 64 chunks of 16B
      #pragma unroll
      for (int j = 0; j < 4; ++j) {
        int c = tid + j * NTHR;
        int r = c >> 6, col = c & 63;
        async_g2l_b128(sX_off + (unsigned)(r * (SX_STRIDE * 2) + col * 16),
                       xqt + (size_t)(m0 + r) * I_ + col * 8);
      }
      wait_async0();
      __syncthreads();
    }

    // ---- phase 1: h_new tile = hq @ W_rec^T + b_rec + xq_t @ W_in^T ----
    {
      v8f acc = {};
      const _Float16* aP = sA + (size_t)l15 * SA_STRIDE + kb;  // LDS A row
      const _Float16* bP = w16rec + (size_t)bN * H_;
      #pragma unroll 4
      for (int k = 0; k < H_; k += 32) {
        v8h lo = *(const v8h*)(aP + k);
        v8h hi = *(const v8h*)(aP + k + 8);
        v16h a;
        #pragma unroll
        for (int i = 0; i < 8; ++i) { a[i] = lo[i]; a[i + 8] = hi[i]; }
        v16h b = *(const v16h*)(bP + k + kB);
        acc = __builtin_amdgcn_wmma_f32_16x16x32_f16(false, a, false, b, (short)0, acc, false, false);
      }
      const _Float16* axP = sX + (size_t)l15 * SX_STRIDE + kb;
      const _Float16* bwP = w16in + (size_t)bN * I_;
      #pragma unroll 4
      for (int k = 0; k < I_; k += 32) {
        v8h lo = *(const v8h*)(axP + k);
        v8h hi = *(const v8h*)(axP + k + 8);
        v16h a;
        #pragma unroll
        for (int i = 0; i < 8; ++i) { a[i] = lo[i]; a[i + 8] = hi[i]; }
        v16h b = *(const v16h*)(bwP + k + kB);
        acc = __builtin_amdgcn_wmma_f32_16x16x32_f16(false, a, false, b, (short)0, acc, false, false);
      }
      float tmn = __builtin_inff(), tmx = -__builtin_inff();
      #pragma unroll
      for (int r = 0; r < 8; ++r) {
        float v = acc[r] + brv;
        hnx[(size_t)(mBase + r) * H_ + bN] = v;
        tmn = fminf(tmn, v);
        tmx = fmaxf(tmx, v);
      }
      for (int o = 16; o; o >>= 1) {
        tmn = fminf(tmn, __shfl_xor(tmn, o, 32));
        tmx = fmaxf(tmx, __shfl_xor(tmx, o, 32));
      }
      if (lane == 0) {
        atomicMin(hmm + 2 * nxt,     fenc(tmn));
        atomicMax(hmm + 2 * nxt + 1, fenc(tmx));
      }
    }
    gbar(bar, bar + 1, ph);

    // ---- phase 2: y_t tile = h_new @ W_out^T + b_out (waves 0..63) ----
    if (gwave < 64) {
      v8f acc = {};
      const float*    haP = hnx + (size_t)aRowY * H_;
      const _Float16* bwP = w16out + (size_t)bNY * H_;
      #pragma unroll 2
      for (int k = 0; k < H_; k += 32) {
        v16h a;
        const float* pa = haP + k + kb;
        #pragma unroll
        for (int i = 0; i < 8; ++i) { a[i] = (_Float16)pa[i]; a[i + 8] = (_Float16)pa[8 + i]; }
        v16h b = *(const v16h*)(bwP + k + kB);
        acc = __builtin_amdgcn_wmma_f32_16x16x32_f16(false, a, false, b, (short)0, acc, false, false);
      }
      float bo = bout[bNY];
      #pragma unroll
      for (int r = 0; r < 8; ++r) {
        int bb = mBaseY + r;
        dout[(size_t)bb * T_ * O_ + (size_t)t * O_ + bNY] = acc[r] + bo;
      }
    }
    // next phase 0 only reads data produced before the last barrier -> no 3rd barrier
  }
}

// ---------------- host launcher ----------------
extern "C" void kernel_launch(void* const* d_in, const int* in_sizes, int n_in,
                              void* d_out, int out_size, void* d_ws, size_t ws_size,
                              hipStream_t stream) {
  const float* x    = (const float*)d_in[0];  // [B,T,I]
  const float* Win  = (const float*)d_in[1];  // [H,I]
  const float* Wrec = (const float*)d_in[2];  // [H,H]
  const float* brec = (const float*)d_in[3];  // [H]
  const float* Wout = (const float*)d_in[4];  // [O,H]
  const float* bout = (const float*)d_in[5];  // [O]
  float* y = (float*)d_out;

  char* ws = (char*)d_ws;
  size_t off = 0;
  auto carve = [&](size_t bytes) -> void* {
    off = (off + 255) & ~(size_t)255;
    void* p = ws + off;
    off += bytes;
    return p;
  };
  _Float16* w16rec = (_Float16*)carve((size_t)H_ * H_ * 2);
  _Float16* w16in  = (_Float16*)carve((size_t)H_ * I_ * 2);
  _Float16* w16out = (_Float16*)carve((size_t)O_ * H_ * 2);
  _Float16* xq16   = (_Float16*)carve((size_t)T_ * B_ * I_ * 2);
  _Float16* hq     = (_Float16*)carve((size_t)B_ * H_ * 2);
  float*    hb0    = (float*)carve((size_t)B_ * H_ * 4);
  float*    hb1    = (float*)carve((size_t)B_ * H_ * 4);
  float*    xmm    = (float*)carve((size_t)T_ * 2 * 4);
  unsigned* hmm    = (unsigned*)carve(4 * 4);
  unsigned* bar    = (unsigned*)carve(2 * 4);

  k_cvt<<<(H_ * H_ + 255) / 256, 256, 0, stream>>>(Wrec, w16rec, H_ * H_);
  k_cvt<<<(H_ * I_ + 255) / 256, 256, 0, stream>>>(Win,  w16in,  H_ * I_);
  k_cvt<<<(O_ * H_ + 255) / 256, 256, 0, stream>>>(Wout, w16out, O_ * H_);
  k_xmm<<<T_, 256, 0, stream>>>(x, xmm);
  long long nq = (long long)B_ * T_ * I_;
  k_xq<<<(unsigned)((nq + 255) / 256), 256, 0, stream>>>(x, xmm, xq16);
  k_init<<<(B_ * H_ + 255) / 256, 256, 0, stream>>>(hb0, hmm, bar);
  k_main<<<NWG, NTHR, SMEM_BYTES, stream>>>(w16rec, w16in, w16out, xq16, brec, bout,
                                            hq, hb0, hb1, hmm, bar, y);
}
